// CommunicationNet_62380105008228
// MI455X (gfx1250) — compile-verified, matrix operand order checked by reading, not run on hardware
//
#include <hip/hip_runtime.h>

// Problem constants (match the reference).
#define RR 16
#define TT 512
#define NN 256
#define DD 22    // sensing dim
#define HH 22    // hidden width
#define PCOL 32  // padded PRE row width (f16 elements) -> 64B rows

typedef __attribute__((ext_vector_type(16))) _Float16 v16h;
typedef __attribute__((ext_vector_type(8)))  float    v8f;

// ---------------------------------------------------------------------------
// Phase 1: PRE[m, j] = runs[m, :] @ W1[0:22, j] + b1[j]   (m = r*T*N + t*N + i)
// One wave per 16-row tile; v_wmma_f32_16x16x32_f16, K padded 22->32 in LDS,
// two 16-col tiles cover cols 0..31 (22 valid + zero padding). All fragment
// reads and D stores are guard-free thanks to zero-padded staging.
// ---------------------------------------------------------------------------
__global__ void __launch_bounds__(256)
comm_pre_gemm(const float* __restrict__ runs,  // [M, 22]
              const float* __restrict__ W1,    // [24, 22]
              const float* __restrict__ b1,    // [22]
              _Float16* __restrict__ pre,      // [M, PCOL] (f16, zero-padded)
              long Mtiles) {
  __shared__ float Wpad[32][33];      // K x col, zero-padded, +1 col anti-conflict
  __shared__ float b1pad[32];
  __shared__ float Apad[8][16][33];   // per-wave 16-row staging, zero-padded

  const int tid   = threadIdx.x;
  const int wslot = tid >> 5;
  const int lane  = tid & 31;
  const long wave = ((long)blockIdx.x * (blockDim.x >> 5)) + wslot;
  const bool activeWave = (wave < Mtiles);

  // ---- Block-wide: fill Wpad/b1pad (clamped loads + select; no exec branching)
  for (int idx = tid; idx < 32 * 32; idx += 256) {
    const int k = idx >> 5, c = idx & 31;
    const int kc = (k < DD) ? k : (DD - 1);
    const int cc = (c < HH) ? c : (HH - 1);
    float v = W1[kc * HH + cc];               // always-valid address
    if (k >= DD || c >= HH) v = 0.0f;         // v_cndmask, not a branch
    Wpad[k][c] = v;
  }
  if (tid < 32) b1pad[tid] = (tid < HH) ? b1[tid] : 0.0f;

  // ---- Per-wave: zero Apad, then coalesced fill of the 16x22 valid region
#pragma unroll 4
  for (int idx = lane; idx < 16 * 32; idx += 32)
    Apad[wslot][idx >> 5][idx & 31] = 0.0f;

  const long rowbase = wave * 16;
  if (activeWave) {
    const float* abase = runs + rowbase * DD;
#pragma unroll
    for (int idx = lane; idx < 16 * DD; idx += 32)
      Apad[wslot][idx / DD][idx % DD] = abase[idx];   // coalesced 128B/iter
  }
  __syncthreads();

  if (activeWave) {
    const int mrow  = lane & 15;
    const int kbase = (lane < 16) ? 0 : 8;  // ISA 16-bit A/B lane->K mapping

    // ---- A fragment: 16x32 f16, row = mrow, halves h: K = kbase + (h<8 ? h : 8+h)
    v16h a;
#pragma unroll
    for (int h = 0; h < 16; ++h) {
      const int k = kbase + ((h < 8) ? h : (8 + h));
      a[h] = (_Float16)Apad[wslot][mrow][k];
    }

#pragma unroll
    for (int nt = 0; nt < 2; ++nt) {
      const int col = nt * 16 + (lane & 15);

      // ---- B fragment: 32x16 f16, column = col, same per-lane K mapping
      v16h b;
#pragma unroll
      for (int h = 0; h < 16; ++h) {
        const int k = kbase + ((h < 8) ? h : (8 + h));
        b[h] = (_Float16)Wpad[k][col];
      }

      // ---- C init: bias broadcast down the output column
      const float bias = b1pad[col];
      v8f c;
#pragma unroll
      for (int v = 0; v < 8; ++v) c[v] = bias;

      c = __builtin_amdgcn_wmma_f32_16x16x32_f16(
          /*neg_a=*/false, a, /*neg_b=*/false, b,
          /*c_mod=*/(short)0, c, /*reuse_a=*/false, /*reuse_b=*/false);

      // ---- D store: element (M = v + 8*(lane>=16), N = col); guard-free
      const long rbase = rowbase + ((lane < 16) ? 0 : 8);
#pragma unroll
      for (int v = 0; v < 8; ++v)
        pre[(rbase + v) * PCOL + col] = (_Float16)c[v];
    }
  }
}

// ---------------------------------------------------------------------------
// Phase 2: serial recurrence. One wave per run (16 waves total).
// Lane j (< 22) owns hidden unit j; comm[258] lives in LDS; the 22->2
// output dot products are wave-level shfl_xor reduction trees.
// ---------------------------------------------------------------------------
__global__ void comm_recurrence(const _Float16* __restrict__ pre,   // [M, PCOL]
                                const float* __restrict__ comm0,    // [R, 258]
                                const float* __restrict__ W1,       // [24, 22]
                                const float* __restrict__ W2,       // [22, 2]
                                const float* __restrict__ b2,       // [2]
                                float* __restrict__ out) {          // [R, T, N]
  __shared__ float comm[NN + 2];
  const int r = blockIdx.x;
  const int j = threadIdx.x;           // 0..31, one wave32 per block

  // init comm with zero sentinels at both ends
  for (int idx = j; idx < NN + 2; idx += 32) {
    float v = comm0[(long)r * (NN + 2) + idx];
    if (idx == 0 || idx == NN + 1) v = 0.0f;
    comm[idx] = v;
  }
  __syncthreads();

  const bool  act = (j < HH);
  const float wl  = act ? W1[DD * HH + j]       : 0.0f;  // W1 row 22 (left)
  const float wr  = act ? W1[(DD + 1) * HH + j] : 0.0f;  // W1 row 23 (right)
  const float w20 = act ? W2[j * 2 + 0] : 0.0f;
  const float w21 = act ? W2[j * 2 + 1] : 0.0f;
  const float bias0 = b2[0];
  const float bias1 = b2[1];

  const _Float16* prebase = pre + (long)r * TT * NN * PCOL;
  float*          outbase = out + (long)r * TT * NN;

  for (int t = 0; t < TT; ++t) {
    for (int i = 0; i < NN; ++i) {
      const long idx = (long)t * NN + i;

      // one aligned 64B row of PRE; lanes >= 22 read padding zeros
      const float p     = (float)prebase[idx * PCOL + j];
      const float left  = comm[i];        // updated this timestep by agent i-1
      const float right = comm[i + 2];    // stale from previous timestep

      const float h = act ? tanhf(fmaf(left, wl, fmaf(right, wr, p))) : 0.0f;

      float s0 = h * w20;
      float s1 = h * w21;
#pragma unroll
      for (int off = 16; off > 0; off >>= 1) {
        s0 += __shfl_xor(s0, off, 32);
        s1 += __shfl_xor(s1, off, 32);
      }

      if (j == 0) {
        const float z  = s1 + bias1;
        comm[i + 1]    = 1.0f / (1.0f + __expf(-z));   // sigmoid(out[1])
        outbase[idx]   = s0 + bias0;                    // control (out[0])
      }
      __syncthreads();  // single-wave barrier: orders the LDS write for i+1
    }
  }
}

// ---------------------------------------------------------------------------
// Launch. Inputs (setup_inputs order): runs, comm0, W1, b1, W2, b2.
// Workspace holds PRE as f16, rows padded to 32: R*T*N*32*2 = 128 MB.
// ---------------------------------------------------------------------------
extern "C" void kernel_launch(void* const* d_in, const int* in_sizes, int n_in,
                              void* d_out, int out_size, void* d_ws, size_t ws_size,
                              hipStream_t stream) {
  const float* runs  = (const float*)d_in[0];
  const float* comm0 = (const float*)d_in[1];
  const float* W1    = (const float*)d_in[2];
  const float* b1    = (const float*)d_in[3];
  const float* W2    = (const float*)d_in[4];
  const float* b2    = (const float*)d_in[5];
  float*       out   = (float*)d_out;

  _Float16* pre = (_Float16*)d_ws;  // [R*T*N, PCOL] f16

  const long M      = (long)RR * TT * NN;   // 2,097,152 rows
  const long Mtiles = M / 16;               // 131,072 row tiles (one wave each)

  // Phase 1: 8 waves / block of 256 threads.
  const int threads = 256;
  const int wavesPerBlock = threads / 32;
  const int blocks = (int)((Mtiles + wavesPerBlock - 1) / wavesPerBlock);
  comm_pre_gemm<<<blocks, threads, 0, stream>>>(runs, W1, b1, pre, Mtiles);

  // Phase 2: one wave32 per run.
  comm_recurrence<<<RR, 32, 0, stream>>>(pre, comm0, W1, W2, b2, out);
}
